// MultiHeadAttention_77927886619157
// MI455X (gfx1250) — matrix operand order repaired and checked by reading, compiled, not run on hardware
//
#include <hip/hip_runtime.h>
#include <stdint.h>

#define B_  2
#define T_  2048
#define E_  1024
#define H_  8
#define D_  512
#define HD_ (H_ * D_)   // 4096
#define M_  (B_ * T_)   // 4096

typedef __attribute__((ext_vector_type(16))) __bf16 v16bf;
typedef __attribute__((ext_vector_type(8)))  float  v8f;
typedef __attribute__((ext_vector_type(4)))  unsigned int v4u;
typedef __attribute__((ext_vector_type(8)))  int v8i_t;
typedef __attribute__((ext_vector_type(4)))  int v4i_t;

union FragBF {
    v16bf v;
    uint32_t u[8];
    uint4 q[2];
};
static_assert(sizeof(FragBF) == 32, "frag must be 8 VGPRs");

__device__ __forceinline__ uint16_t f2bf(float f) {
    uint32_t u = __float_as_uint(f);
    u += 0x7FFFu + ((u >> 16) & 1u);   // round to nearest even
    return (uint16_t)(u >> 16);
}
__device__ __forceinline__ float bf2f(uint16_t h) {
    return __uint_as_float(((uint32_t)h) << 16);
}

__device__ __forceinline__ v8f wmma_bf16(const FragBF& a, const FragBF& b, v8f c) {
    // D = A(16x32 bf16) * B(32x16 bf16) + C(16x16 f32)
    return __builtin_amdgcn_wmma_f32_16x16x32_bf16(false, a.v, false, b.v,
                                                   (short)0, c, false, false);
}

// ---------------------------------------------------------------------------
// fp32 -> bf16 elementwise convert
// ---------------------------------------------------------------------------
__global__ __launch_bounds__(256) void convert_f32_bf16(
    const float* __restrict__ src, uint16_t* __restrict__ dst, int n) {
    int i = blockIdx.x * 256 + threadIdx.x;
    if (i < n) dst[i] = f2bf(src[i]);
}

// ---------------------------------------------------------------------------
// C[M,N] = A[M,K] @ W[N,K]^T      (A, W bf16 row-major; C f32 or bf16)
// Block: 256 threads (8 waves), 128x128 tile.
// Double-buffered LDS K-staging (BK=32): next tile is fetched to registers
// during compute, stored to the alternate buffer -> one barrier per K-step.
// Inner fragment loop is software-pipelined (load B[f+1] before WMMA f).
// ---------------------------------------------------------------------------
template <int STORE_BF16>
__global__ __launch_bounds__(256) void gemm_bf16_wmma(
    const uint16_t* __restrict__ A,
    const uint16_t* __restrict__ W,
    void* __restrict__ Cout,
    int M, int N, int K)
{
    constexpr int LDT = 40;  // LDS row pitch in bf16 (80B): conflict-free b128 frag reads
    __shared__ __align__(16) uint16_t At[2][128 * LDT];
    __shared__ __align__(16) uint16_t Bt[2][128 * LDT];

    const int tid  = threadIdx.x;
    const int wave = tid >> 5;
    const int lane = tid & 31;
    const int half = lane >> 4;
    const int mn   = lane & 15;
    const int m0   = blockIdx.x * 128;
    const int n0   = blockIdx.y * 128;

    // this thread's two 16B staging chunks: rows r0 and r0+64, chunk col c0
    const int r0 = tid >> 2;
    const int c0 = tid & 3;

    v8f acc[8];
#pragma unroll
    for (int f = 0; f < 8; ++f)
#pragma unroll
        for (int r = 0; r < 8; ++r) acc[f][r] = 0.f;

    // stage K-slice 0 into buffer 0
    {
        *reinterpret_cast<uint4*>(&At[0][r0 * LDT + c0 * 8]) =
            *reinterpret_cast<const uint4*>(A + (size_t)(m0 + r0) * K + c0 * 8);
        *reinterpret_cast<uint4*>(&At[0][(r0 + 64) * LDT + c0 * 8]) =
            *reinterpret_cast<const uint4*>(A + (size_t)(m0 + r0 + 64) * K + c0 * 8);
        *reinterpret_cast<uint4*>(&Bt[0][r0 * LDT + c0 * 8]) =
            *reinterpret_cast<const uint4*>(W + (size_t)(n0 + r0) * K + c0 * 8);
        *reinterpret_cast<uint4*>(&Bt[0][(r0 + 64) * LDT + c0 * 8]) =
            *reinterpret_cast<const uint4*>(W + (size_t)(n0 + r0 + 64) * K + c0 * 8);
    }
    __syncthreads();

    for (int k0 = 0; k0 < K; k0 += 32) {
        const int cur  = (k0 >> 5) & 1;
        const int nxt  = cur ^ 1;
        const bool more = (k0 + 32) < K;

        // prefetch next K-slice into registers (overlaps with WMMAs below)
        uint4 ra0, ra1, rb0, rb1;
        if (more) {
            const int k1 = k0 + 32;
            ra0 = *reinterpret_cast<const uint4*>(A + (size_t)(m0 + r0) * K + k1 + c0 * 8);
            ra1 = *reinterpret_cast<const uint4*>(A + (size_t)(m0 + r0 + 64) * K + k1 + c0 * 8);
            rb0 = *reinterpret_cast<const uint4*>(W + (size_t)(n0 + r0) * K + k1 + c0 * 8);
            rb1 = *reinterpret_cast<const uint4*>(W + (size_t)(n0 + r0 + 64) * K + k1 + c0 * 8);
        }

        // A fragment (16x32): lane m=mn; dword j: K = half*16 + 2j (16 contiguous)
        FragBF a;
        const uint16_t* ap = &At[cur][(wave * 16 + mn) * LDT + half * 16];
        a.q[0] = *reinterpret_cast<const uint4*>(ap);
        a.q[1] = *reinterpret_cast<const uint4*>(ap + 8);

        // B fragments (32x16): lane n=mn; dword j: K = half*16 + 2j (16 contiguous)
        const uint16_t* bbase = &Bt[cur][mn * LDT + half * 16];
        FragBF bcur;
        bcur.q[0] = *reinterpret_cast<const uint4*>(bbase);
        bcur.q[1] = *reinterpret_cast<const uint4*>(bbase + 8);
#pragma unroll
        for (int f = 0; f < 7; ++f) {
            FragBF bnext;
            const uint16_t* bp = bbase + (f + 1) * 16 * LDT;
            bnext.q[0] = *reinterpret_cast<const uint4*>(bp);
            bnext.q[1] = *reinterpret_cast<const uint4*>(bp + 8);
            acc[f] = wmma_bf16(a, bcur, acc[f]);
            bcur = bnext;
        }
        acc[7] = wmma_bf16(a, bcur, acc[7]);

        // store prefetched slice into the other buffer; single barrier per step
        if (more) {
            *reinterpret_cast<uint4*>(&At[nxt][r0 * LDT + c0 * 8]) = ra0;
            *reinterpret_cast<uint4*>(&At[nxt][(r0 + 64) * LDT + c0 * 8]) = ra1;
            *reinterpret_cast<uint4*>(&Bt[nxt][r0 * LDT + c0 * 8]) = rb0;
            *reinterpret_cast<uint4*>(&Bt[nxt][(r0 + 64) * LDT + c0 * 8]) = rb1;
            __syncthreads();
        }
    }

    // C/D layout: lane col n=mn; VGPR r -> row r + 8*half
#pragma unroll
    for (int f = 0; f < 8; ++f)
#pragma unroll
        for (int r = 0; r < 8; ++r) {
            const int row = m0 + wave * 16 + r + 8 * half;
            const int col = n0 + f * 16 + mn;
            if (STORE_BF16)
                ((uint16_t*)Cout)[(size_t)row * N + col] = f2bf(acc[f][r]);
            else
                ((float*)Cout)[(size_t)row * N + col] = acc[f][r];
        }
}

// ---------------------------------------------------------------------------
// RoPE on Q,K (bf16 in/out) + V transpose to [B,H,D,T] bf16.
// ---------------------------------------------------------------------------
__global__ __launch_bounds__(256) void rope_convert(
    const uint16_t* __restrict__ q0, const uint16_t* __restrict__ k0,
    const uint16_t* __restrict__ v0,
    uint16_t* __restrict__ qr, uint16_t* __restrict__ kr,
    uint16_t* __restrict__ vt)
{
    const int DH = D_ / 2;
    size_t i = (size_t)blockIdx.x * 256 + threadIdx.x;  // < B*T*H*DH
    const int p = (int)(i % DH);
    size_t r = i / DH;
    const int h = (int)(r % H_); r /= H_;
    const int t = (int)(r % T_);
    const int b = (int)(r / T_);
    const size_t base = (size_t)(b * T_ + t) * HD_ + h * D_ + 2 * p;

    const float inv_freq = powf(10000.f, -((float)(2 * p) / (float)D_));
    const float ang = (float)t * inv_freq;
    const float c = cosf(ang), s = sinf(ang);

    const float q1 = bf2f(q0[base]), q2 = bf2f(q0[base + 1]);
    qr[base]     = f2bf(q1 * c - q2 * s);
    qr[base + 1] = f2bf(q1 * s + q2 * c);

    const float k1 = bf2f(k0[base]), k2 = bf2f(k0[base + 1]);
    kr[base]     = f2bf(k1 * c - k2 * s);
    kr[base + 1] = f2bf(k1 * s + k2 * c);

    const size_t vtb = ((size_t)((b * H_ + h) * D_ + 2 * p)) * T_ + t;
    vt[vtb]      = v0[base];
    vt[vtb + T_] = v0[base + 1];
}

// ---------------------------------------------------------------------------
// Flash-style causal attention. Block = (b, h, 32 query rows), 256 threads.
// V key-block (512 x 64 bf16) is staged into LDS by the Tensor Data Mover
// (one descriptor per key block, issued by wave 0, overlapped with QK^T),
// with D# padding giving a 72-element row pitch (conflict-free ds_load_b128).
// Fragment loops are software-pipelined.
// ---------------------------------------------------------------------------
#define QB   32
#define KBLK 64
#define VPITCH 72   // 64 data + 8 pad bf16 per row (144B)

#define HAVE_TDM __has_builtin(__builtin_amdgcn_tensor_load_to_lds)

__global__ __launch_bounds__(256) void attention_wmma(
    const uint16_t* __restrict__ qb,  // [B*T, HD]
    const uint16_t* __restrict__ kb,  // [B*T, HD]
    const uint16_t* __restrict__ vt,  // [B*H*D, T]
    uint16_t* __restrict__ ao)        // [B*T, HD]
{
    __shared__ __align__(16) uint16_t VtL[D_ * VPITCH];   // 73.7 KB
    __shared__ __align__(16) float    Sbuf[QB * KBLK];
    __shared__ __align__(16) uint16_t Pbuf[QB * KBLK];
    __shared__ float mrow[QB], lrow[QB], arow[QB];

    const int tid  = threadIdx.x;
    const int wave = tid >> 5;
    const int lane = tid & 31;
    const int half = lane >> 4;
    const int mn   = lane & 15;
    const int qs   = wave >> 2;   // 0..1
    const int quad = wave & 3;    // 0..3

    const int qBase = blockIdx.x * QB;
    const int h     = blockIdx.y;
    const int b     = blockIdx.z;
    const float scale = 0.044194173824159216f;  // 1/sqrt(512)

    v8f acc[8];
#pragma unroll
    for (int f = 0; f < 8; ++f)
#pragma unroll
        for (int r = 0; r < 8; ++r) acc[f][r] = 0.f;

    if (tid < QB) { mrow[tid] = -1e30f; lrow[tid] = 0.f; }
    __syncthreads();

    const size_t qrow0 = (size_t)(b * T_ + qBase + qs * 16 + mn) * HD_ + h * D_;
    const uint16_t* vsrc0 = vt + (size_t)((b * H_ + h) * D_) * T_;  // [512, T] slice
    const int nkb = (qBase + QB + KBLK - 1) / KBLK;

    for (int kblk = 0; kblk < nkb; ++kblk) {
        const int kBase = kblk * KBLK;

        // ---- Issue V-tile DMA: global [512 x 64] (pitch T) -> LDS (pitch 72)
#if HAVE_TDM
        if (wave == 0) {
            const uint64_t gaddr = (uint64_t)(uintptr_t)(const void*)(vsrc0 + kBase);
            const uint32_t ldsa  = (uint32_t)(uintptr_t)(void*)&VtL[0];
            v4u g0;
            g0[0] = 1u;                           // count=1, load descriptor
            g0[1] = ldsa;                         // lds_addr
            g0[2] = (uint32_t)gaddr;              // global_addr[31:0]
            g0[3] = (uint32_t)(gaddr >> 32) | (2u << 30);  // addr[56:32] | type=2
            v8i_t g1;
            g1[0] = (int)((1u << 16)      // data_size = 2B
                        | (1u << 20)      // pad_enable
                        | (4u << 22)      // pad_interval: 32 dwords (128B rows)
                        | (3u << 25));    // pad_amount: 4 dwords (8 bf16)
            g1[1] = (int)((uint32_t)T_ << 16);             // tensor_dim0 = 2048
            g1[2] = (int)((uint32_t)(B_ * H_ * D_) << 16); // tensor_dim1
            g1[3] = (int)((uint32_t)KBLK << 16);           // tile_dim0 = 64
            g1[4] = D_;                                    // tile_dim1 = 512
            g1[5] = T_;                                    // tensor_dim0_stride
            g1[6] = 0;
            g1[7] = 0;
            v4i_t gz;
            gz[0] = 0; gz[1] = 0; gz[2] = 0; gz[3] = 0;
            v8i_t gz8;
            gz8[0] = 0; gz8[1] = 0; gz8[2] = 0; gz8[3] = 0;
            gz8[4] = 0; gz8[5] = 0; gz8[6] = 0; gz8[7] = 0;
            __builtin_amdgcn_tensor_load_to_lds(g0, g1, gz, gz, gz8, 0);
        }
#else
        for (int i = tid; i < D_ * (KBLK / 8); i += 256) {
            const int r = i >> 3, c = i & 7;
            *reinterpret_cast<uint4*>(&VtL[r * VPITCH + c * 8]) =
                *reinterpret_cast<const uint4*>(vsrc0 + (size_t)r * T_ + kBase + c * 8);
        }
#endif

        // ---- Phase 1: S[qs rows][quad keys] = Q . K^T over D=512 (pipelined) ----
        v8f s;
#pragma unroll
        for (int r = 0; r < 8; ++r) s[r] = 0.f;
        const size_t krow = (size_t)(b * T_ + kBase + quad * 16 + mn) * HD_ + h * D_;

        FragBF aq, bk;
        aq.q[0] = *reinterpret_cast<const uint4*>(qb + qrow0 + half * 16);
        aq.q[1] = *reinterpret_cast<const uint4*>(qb + qrow0 + half * 16 + 8);
        bk.q[0] = *reinterpret_cast<const uint4*>(kb + krow + half * 16);
        bk.q[1] = *reinterpret_cast<const uint4*>(kb + krow + half * 16 + 8);
        for (int dk = 0; dk < D_ - 32; dk += 32) {
            FragBF an, bn;
            an.q[0] = *reinterpret_cast<const uint4*>(qb + qrow0 + dk + 32 + half * 16);
            an.q[1] = *reinterpret_cast<const uint4*>(qb + qrow0 + dk + 32 + half * 16 + 8);
            bn.q[0] = *reinterpret_cast<const uint4*>(kb + krow + dk + 32 + half * 16);
            bn.q[1] = *reinterpret_cast<const uint4*>(kb + krow + dk + 32 + half * 16 + 8);
            s = wmma_bf16(aq, bk, s);
            aq = an; bk = bn;
        }
        s = wmma_bf16(aq, bk, s);

#pragma unroll
        for (int r = 0; r < 8; ++r)
            Sbuf[(qs * 16 + r + 8 * half) * KBLK + quad * 16 + mn] = s[r] * scale;
        __syncthreads();

        // ---- Phase 2: online softmax (wave 0, one row per lane) ----
        if (wave == 0) {
            const int row = tid;
            const int qg  = qBase + row;
            const float mo = mrow[row];
            float mx = mo;
            for (int j = 0; j < KBLK; ++j) {
                const float sv = (kBase + j <= qg) ? Sbuf[row * KBLK + j] : -1e9f;
                mx = fmaxf(mx, sv);
            }
            const float al = __expf(mo - mx);
            float sum = 0.f;
            for (int j = 0; j < KBLK; ++j) {
                const float sv = (kBase + j <= qg) ? Sbuf[row * KBLK + j] : -1e9f;
                const float pv = __expf(sv - mx);
                sum += pv;
                Pbuf[row * KBLK + j] = f2bf(pv);
            }
            mrow[row] = mx;
            lrow[row] = lrow[row] * al + sum;
            arow[row] = al;
#if HAVE_TDM
            __builtin_amdgcn_s_wait_tensorcnt(0);  // V tile resident before release
#endif
        }
        __syncthreads();

        // ---- Phase 3: rescale O, then O += P @ V (V from LDS, pipelined) ----
#pragma unroll
        for (int r = 0; r < 8; ++r) {
            const float al = arow[qs * 16 + r + 8 * half];
#pragma unroll
            for (int f = 0; f < 8; ++f) acc[f][r] *= al;
        }
#pragma unroll
        for (int kk = 0; kk < 2; ++kk) {
            FragBF a;
            const uint16_t* pp = &Pbuf[(qs * 16 + mn) * KBLK + kk * 32 + half * 16];
            a.q[0] = *reinterpret_cast<const uint4*>(pp);
            a.q[1] = *reinterpret_cast<const uint4*>(pp + 8);

            const uint16_t* vbase = &VtL[(quad * 128 + mn) * VPITCH + kk * 32 + half * 16];
            FragBF bcur;
            bcur.q[0] = *reinterpret_cast<const uint4*>(vbase);
            bcur.q[1] = *reinterpret_cast<const uint4*>(vbase + 8);
#pragma unroll
            for (int f = 0; f < 7; ++f) {
                FragBF bnext;
                const uint16_t* vp = vbase + (f + 1) * 16 * VPITCH;
                bnext.q[0] = *reinterpret_cast<const uint4*>(vp);
                bnext.q[1] = *reinterpret_cast<const uint4*>(vp + 8);
                acc[kk == 0 ? f : f] = wmma_bf16(a, bcur, acc[f]);
                bcur = bnext;
            }
            acc[7] = wmma_bf16(a, bcur, acc[7]);
        }
        // keep next iteration's V-tile DMA from overwriting VtL while phase 3
        // readers are still in flight
        __syncthreads();
    }

    // ---- Epilogue: normalize by row sum, store bf16 for final projection ----
#pragma unroll
    for (int r = 0; r < 8; ++r) {
        const int row = qs * 16 + r + 8 * half;
        const float inv = 1.f / lrow[row];
        const size_t orow = (size_t)(b * T_ + qBase + row) * HD_ + h * D_ + quad * 128;
#pragma unroll
        for (int f = 0; f < 8; ++f)
            ao[orow + f * 16 + mn] = f2bf(acc[f][r] * inv);
    }
}

// ---------------------------------------------------------------------------
extern "C" void kernel_launch(void* const* d_in, const int* in_sizes, int n_in,
                              void* d_out, int out_size, void* d_ws, size_t ws_size,
                              hipStream_t stream) {
    (void)in_sizes; (void)n_in; (void)out_size; (void)ws_size;

    const float* x  = (const float*)d_in[0];
    const float* wq = (const float*)d_in[1];
    const float* wk = (const float*)d_in[2];
    const float* wv = (const float*)d_in[3];
    const float* wo = (const float*)d_in[4];

    char* ws = (char*)d_ws;
    size_t off = 0;
    auto alloc = [&](size_t bytes) {
        char* p = ws + off;
        off += (bytes + 255) & ~(size_t)255;
        return p;
    };

    const size_t nXE = (size_t)M_ * E_;    // 4.19M elems
    const size_t nW  = (size_t)HD_ * E_;   // 4.19M elems
    const size_t nQ  = (size_t)M_ * HD_;   // 16.8M elems

    uint16_t* xb  = (uint16_t*)alloc(nXE * 2);
    uint16_t* wqb = (uint16_t*)alloc(nW * 2);
    uint16_t* wkb = (uint16_t*)alloc(nW * 2);
    uint16_t* wvb = (uint16_t*)alloc(nW * 2);
    uint16_t* wob = (uint16_t*)alloc(nW * 2);
    uint16_t* q0  = (uint16_t*)alloc(nQ * 2);
    uint16_t* k0  = (uint16_t*)alloc(nQ * 2);
    uint16_t* v0  = (uint16_t*)alloc(nQ * 2);
    uint16_t* qr  = (uint16_t*)alloc(nQ * 2);
    uint16_t* kr  = (uint16_t*)alloc(nQ * 2);
    uint16_t* vtr = (uint16_t*)alloc(nQ * 2);
    uint16_t* aob = (uint16_t*)alloc(nQ * 2);

    const dim3 blk(256);

    convert_f32_bf16<<<(unsigned)(nXE / 256), blk, 0, stream>>>(x,  xb,  (int)nXE);
    convert_f32_bf16<<<(unsigned)(nW  / 256), blk, 0, stream>>>(wq, wqb, (int)nW);
    convert_f32_bf16<<<(unsigned)(nW  / 256), blk, 0, stream>>>(wk, wkb, (int)nW);
    convert_f32_bf16<<<(unsigned)(nW  / 256), blk, 0, stream>>>(wv, wvb, (int)nW);
    convert_f32_bf16<<<(unsigned)(nW  / 256), blk, 0, stream>>>(wo, wob, (int)nW);

    // QKV projections: [4096,1024] @ [4096,1024]^T -> bf16 [4096,4096]
    const dim3 gqkv(32, 32);
    gemm_bf16_wmma<1><<<gqkv, blk, 0, stream>>>(xb, wqb, q0, M_, HD_, E_);
    gemm_bf16_wmma<1><<<gqkv, blk, 0, stream>>>(xb, wkb, k0, M_, HD_, E_);
    gemm_bf16_wmma<1><<<gqkv, blk, 0, stream>>>(xb, wvb, v0, M_, HD_, E_);

    // RoPE(Q,K) + V transpose
    rope_convert<<<(unsigned)((size_t)B_ * T_ * H_ * (D_ / 2) / 256), blk, 0, stream>>>(
        q0, k0, v0, qr, kr, vtr);

    // Causal flash attention
    attention_wmma<<<dim3(T_ / QB, H_, B_), blk, 0, stream>>>(qr, kr, vtr, aob);

    // Output projection: [4096,4096] @ [1024,4096]^T -> f32 [4096,1024]
    gemm_bf16_wmma<0><<<dim3(32, 8), blk, 0, stream>>>(aob, wob, d_out, M_, E_, HD_);
}
